// Coembedding_41369124995828
// MI455X (gfx1250) — compile-verified, matrix operand order checked by reading, not run on hardware
//
#include <hip/hip_runtime.h>

// ---------------------------------------------------------------------------
// CDNA5 (gfx1250) co-embedding pipeline:
//   Hm = ReLU(mol @ Wm1^T + bm1) ; mp = Hm @ Wm2^T + bm2 ; mn = rownorm(mp)
//   Hp = ReLU(prot @ Wp1^T + bp1); pp = Hp @ Wp2^T + bp2 ; pn = rownorm(pp)
//   out = (pn @ mn^T) * (1/T)
// All matmuls are NT-GEMMs mapped onto v_wmma_f32_16x16x32_f16 with fp32
// accumulation, double-buffered LDS tiles, and GLOBAL_LOAD_ASYNC_TO_LDS_B128
// for the global->LDS staging.
// ---------------------------------------------------------------------------

typedef _Float16 half8  __attribute__((ext_vector_type(8)));
typedef _Float16 half16 __attribute__((ext_vector_type(16)));
typedef float    f32x8  __attribute__((ext_vector_type(8)));
typedef int      v4i    __attribute__((ext_vector_type(4)));

#define TILE_M 128
#define TILE_N 128
#define TILE_K 32
#define LDS_STRIDE 40   // halves; 80B row pitch keeps ds_read_b128 16B-aligned

#if defined(__has_builtin)
#  if __has_builtin(__builtin_amdgcn_global_load_async_to_lds_b128)
#    define USE_ASYNC_LDS 1
#  endif
#endif
#ifndef USE_ASYNC_LDS
#  define USE_ASYNC_LDS 0
#endif

#if USE_ASYNC_LDS
__device__ __forceinline__ void async_copy16(const _Float16* g, _Float16* l) {
    // Builtin signature (from hipcc diagnostic): (as1 int4*, as3 int4*, int, int).
    // Generic LDS pointer low 32 bits == LDS byte offset (aperture in [63:32]).
    __builtin_amdgcn_global_load_async_to_lds_b128(
        (__attribute__((address_space(1))) v4i*)(size_t)g,
        (__attribute__((address_space(3))) v4i*)(unsigned int)(size_t)l,
        0, 0);
}
__device__ __forceinline__ void wait_async_copies() {
#  if __has_builtin(__builtin_amdgcn_s_wait_asynccnt)
    __builtin_amdgcn_s_wait_asynccnt(0);
#  else
    asm volatile("s_wait_asynccnt 0" ::: "memory");
#  endif
}
#endif

// Load one 16x32 f16 operand fragment from an LDS tile (row-major, stride
// LDS_STRIDE halves) in the ISA 7.12.2 "16-bit A-matrix 16x32" lane layout:
//   lanes 0-15 : row = lane,    K = 0..7   and 16..23
//   lanes 16-31: row = lane-16, K = 8..15  and 24..31
__device__ __forceinline__ half16 load_frag16x32(const _Float16* tile, int lane) {
    const int row = lane & 15;
    const int khi = (lane >> 4) << 3;           // 0 or 8
    const _Float16* p = tile + row * LDS_STRIDE + khi;
    half8 lo = *(const half8*)(p);              // K = khi .. khi+7
    half8 hi = *(const half8*)(p + 16);         // K = khi+16 .. khi+23
    return __builtin_shufflevector(lo, hi, 0, 1, 2, 3, 4, 5, 6, 7,
                                           8, 9, 10, 11, 12, 13, 14, 15);
}

// C[M,N] = A[M,K] * B[N,K]^T  (+bias, ReLU, scale, f16/f32 out)
// Block: 256 threads = 8 waves; 128x128 tile; each wave computes 32x64.
template <bool BIAS, bool RELU, bool OUT_F16>
__global__ __launch_bounds__(256)
void gemm_nt_wmma(const _Float16* __restrict__ A,
                  const _Float16* __restrict__ B,
                  const float* __restrict__ bias,
                  void* __restrict__ Cv,
                  int M, int N, int K,
                  const float* __restrict__ temp_ptr) {
    __shared__ _Float16 As[2][TILE_M * LDS_STRIDE];
    __shared__ _Float16 Bs[2][TILE_N * LDS_STRIDE];

    const int tid  = threadIdx.x;
    const int lane = tid & 31;          // wave32
    const int wave = tid >> 5;          // 0..7
    const int bm0  = blockIdx.y * TILE_M;
    const int bn0  = blockIdx.x * TILE_N;

    // Each thread owns 4 fixed 16B staging chunks per K-tile:
    //   A rows (tid>>2) and (tid>>2)+64, B rows likewise; col = (tid&3)*8.
    const int srow = tid >> 2;                  // 0..63
    const int scol = (tid & 3) << 3;            // 0,8,16,24
    const _Float16* gA0 = A + (size_t)(bm0 + srow) * K + scol;
    const _Float16* gA1 = gA0 + (size_t)64 * K;
    const _Float16* gB0 = B + (size_t)(bn0 + srow) * K + scol;
    const _Float16* gB1 = gB0 + (size_t)64 * K;
    const int ldsOff  = srow * LDS_STRIDE + scol;
    const int ldsOff2 = ldsOff + 64 * LDS_STRIDE;

    // wave sub-tile: 4 waves along M (32 rows each), 2 waves along N (64 cols)
    const int wm = (wave >> 1) * 32;
    const int wn = (wave & 1) * 64;

    f32x8 acc[2][4];
#pragma unroll
    for (int i = 0; i < 2; ++i)
#pragma unroll
        for (int j = 0; j < 4; ++j) acc[i][j] = (f32x8)0.0f;

    const int nkt = K / TILE_K;

    // ---- prologue: stage K-tile 0 into buffer 0 -------------------------
#if USE_ASYNC_LDS
    async_copy16(gA0, &As[0][ldsOff]);
    async_copy16(gA1, &As[0][ldsOff2]);
    async_copy16(gB0, &Bs[0][ldsOff]);
    async_copy16(gB1, &Bs[0][ldsOff2]);
    wait_async_copies();
#else
    {
        half8 ra0 = *(const half8*)gA0;
        half8 ra1 = *(const half8*)gA1;
        half8 rb0 = *(const half8*)gB0;
        half8 rb1 = *(const half8*)gB1;
        *(half8*)&As[0][ldsOff]  = ra0;
        *(half8*)&As[0][ldsOff2] = ra1;
        *(half8*)&Bs[0][ldsOff]  = rb0;
        *(half8*)&Bs[0][ldsOff2] = rb1;
    }
#endif
    __syncthreads();

    // ---- steady state: compute tile kt from buf cur, stage kt+1 to nxt --
    for (int kt = 0; kt < nkt; ++kt) {
        const int cur  = kt & 1;
        const int nxt  = cur ^ 1;
        const int k1   = (kt + 1) * TILE_K;
        const bool more = (kt + 1) < nkt;

#if USE_ASYNC_LDS
        if (more) {
            async_copy16(gA0 + k1, &As[nxt][ldsOff]);
            async_copy16(gA1 + k1, &As[nxt][ldsOff2]);
            async_copy16(gB0 + k1, &Bs[nxt][ldsOff]);
            async_copy16(gB1 + k1, &Bs[nxt][ldsOff2]);
        }
#else
        half8 ra0{}, ra1{}, rb0{}, rb1{};
        if (more) {
            ra0 = *(const half8*)(gA0 + k1);
            ra1 = *(const half8*)(gA1 + k1);
            rb0 = *(const half8*)(gB0 + k1);
            rb1 = *(const half8*)(gB1 + k1);
        }
#endif

        half16 afrag[2], bfrag[4];
#pragma unroll
        for (int i = 0; i < 2; ++i)
            afrag[i] = load_frag16x32(&As[cur][(wm + i * 16) * LDS_STRIDE], lane);
#pragma unroll
        for (int j = 0; j < 4; ++j)
            bfrag[j] = load_frag16x32(&Bs[cur][(wn + j * 16) * LDS_STRIDE], lane);

#pragma unroll
        for (int i = 0; i < 2; ++i)
#pragma unroll
            for (int j = 0; j < 4; ++j)
                acc[i][j] = __builtin_amdgcn_wmma_f32_16x16x32_f16(
                    /*neg_a=*/false, afrag[i], /*neg_b=*/false, bfrag[j],
                    /*c_mod=*/(short)0, acc[i][j],
                    /*reuse_a=*/false, /*reuse_b=*/false);

#if USE_ASYNC_LDS
        wait_async_copies();
#else
        if (more) {
            *(half8*)&As[nxt][ldsOff]  = ra0;
            *(half8*)&As[nxt][ldsOff2] = ra1;
            *(half8*)&Bs[nxt][ldsOff]  = rb0;
            *(half8*)&Bs[nxt][ldsOff2] = rb1;
        }
#endif
        __syncthreads();
    }

    const float scale = temp_ptr ? (1.0f / temp_ptr[0]) : 1.0f;

    // C/D layout (ISA 7.12.2): VGPR g -> M = g + 8*(lane>>4), N = lane&15.
    const int ncol = lane & 15;
    const int mhi  = (lane >> 4) << 3;
#pragma unroll
    for (int i = 0; i < 2; ++i) {
#pragma unroll
        for (int j = 0; j < 4; ++j) {
            const int col = bn0 + wn + j * 16 + ncol;
            const float bv = BIAS ? bias[col] : 0.0f;
#pragma unroll
            for (int g = 0; g < 8; ++g) {
                const int row = bm0 + wm + i * 16 + mhi + g;
                float v = acc[i][j][g] + bv;
                if (RELU) v = fmaxf(v, 0.0f);
                v *= scale;
                const size_t idx = (size_t)row * N + col;
                if (OUT_F16) ((_Float16*)Cv)[idx] = (_Float16)v;
                else         ((float*)Cv)[idx] = v;
            }
        }
    }
}

// fp32 -> fp16 conversion (grid-stride)
__global__ __launch_bounds__(256)
void f32_to_f16_kernel(const float* __restrict__ x, _Float16* __restrict__ y, int n) {
    int i = blockIdx.x * blockDim.x + threadIdx.x;
    const int stride = gridDim.x * blockDim.x;
    for (; i < n; i += stride) y[i] = (_Float16)x[i];
}

// Row-wise L2 normalize with eps clamp: y[r,:] = x[r,:] / max(||x[r,:]||, eps)
__global__ __launch_bounds__(256)
void rownorm_f16_kernel(const float* __restrict__ X, _Float16* __restrict__ Y, int D) {
    __shared__ float red[256];
    const int row = blockIdx.x;
    const float* x = X + (size_t)row * D;
    float ss = 0.0f;
    for (int i = threadIdx.x; i < D; i += 256) {
        const float v = x[i];
        ss += v * v;
    }
    red[threadIdx.x] = ss;
    __syncthreads();
    for (int s = 128; s > 0; s >>= 1) {
        if (threadIdx.x < s) red[threadIdx.x] += red[threadIdx.x + s];
        __syncthreads();
    }
    const float inv = 1.0f / fmaxf(sqrtf(red[0]), 1e-8f);
    _Float16* y = Y + (size_t)row * D;
    for (int i = threadIdx.x; i < D; i += 256) y[i] = (_Float16)(x[i] * inv);
}

extern "C" void kernel_launch(void* const* d_in, const int* in_sizes, int n_in,
                              void* d_out, int out_size, void* d_ws, size_t ws_size,
                              hipStream_t stream) {
    (void)in_sizes; (void)n_in; (void)out_size; (void)ws_size;

    // Problem dims (fixed by reference)
    const int N = 4096, M = 8192, MOL = 768, PROT = 1280, D = 1024;

    const float* molecule = (const float*)d_in[0];
    const float* protein  = (const float*)d_in[1];
    const float* Wm1 = (const float*)d_in[2];
    const float* bm1 = (const float*)d_in[3];
    const float* Wm2 = (const float*)d_in[4];
    const float* bm2 = (const float*)d_in[5];
    const float* Wp1 = (const float*)d_in[6];
    const float* bp1 = (const float*)d_in[7];
    const float* Wp2 = (const float*)d_in[8];
    const float* bp2 = (const float*)d_in[9];
    const float* temperature = (const float*)d_in[10];
    float* out = (float*)d_out;

    // Workspace carve-up (256B aligned regions)
    size_t off = 0;
    auto carve = [&](size_t bytes) {
        void* p = (char*)d_ws + off;
        off += (bytes + 255) & ~(size_t)255;
        return p;
    };
    _Float16* molH  = (_Float16*)carve((size_t)N * MOL * 2);
    _Float16* protH = (_Float16*)carve((size_t)M * PROT * 2);
    _Float16* Wm1H  = (_Float16*)carve((size_t)D * MOL * 2);
    _Float16* Wm2H  = (_Float16*)carve((size_t)D * D * 2);
    _Float16* Wp1H  = (_Float16*)carve((size_t)D * PROT * 2);
    _Float16* Wp2H  = (_Float16*)carve((size_t)D * D * 2);
    _Float16* Hm    = (_Float16*)carve((size_t)N * D * 2);
    _Float16* Hp    = (_Float16*)carve((size_t)M * D * 2);
    float*    mp    = (float*)carve((size_t)N * D * 4);
    float*    pp    = (float*)carve((size_t)M * D * 4);
    _Float16* mn    = (_Float16*)carve((size_t)N * D * 2);
    _Float16* pn    = (_Float16*)carve((size_t)M * D * 2);

    auto cvt = [&](const float* src, _Float16* dst, int n) {
        int blocks = (n + 2047) / 2048;
        if (blocks > 2048) blocks = 2048;
        f32_to_f16_kernel<<<blocks, 256, 0, stream>>>(src, dst, n);
    };
    cvt(molecule, molH,  N * MOL);
    cvt(protein,  protH, M * PROT);
    cvt(Wm1, Wm1H, D * MOL);
    cvt(Wm2, Wm2H, D * D);
    cvt(Wp1, Wp1H, D * PROT);
    cvt(Wp2, Wp2H, D * D);

    // Layer 1: H = ReLU(X @ W1^T + b1), f16 output
    gemm_nt_wmma<true, true, true><<<dim3(D / TILE_N, N / TILE_M), 256, 0, stream>>>(
        molH, Wm1H, bm1, Hm, N, D, MOL, nullptr);
    gemm_nt_wmma<true, true, true><<<dim3(D / TILE_N, M / TILE_M), 256, 0, stream>>>(
        protH, Wp1H, bp1, Hp, M, D, PROT, nullptr);

    // Layer 2: P = H @ W2^T + b2, f32 output (needed in fp32 for norms)
    gemm_nt_wmma<true, false, false><<<dim3(D / TILE_N, N / TILE_M), 256, 0, stream>>>(
        Hm, Wm2H, bm2, mp, N, D, D, nullptr);
    gemm_nt_wmma<true, false, false><<<dim3(D / TILE_N, M / TILE_M), 256, 0, stream>>>(
        Hp, Wp2H, bp2, pp, M, D, D, nullptr);

    // Row normalize (clamped at eps), emit f16 for the final WMMA GEMM
    rownorm_f16_kernel<<<N, 256, 0, stream>>>(mp, mn, D);
    rownorm_f16_kernel<<<M, 256, 0, stream>>>(pp, pn, D);

    // distance = (pn @ mn^T) / temperature, fp32 output [M, N]
    gemm_nt_wmma<false, false, false><<<dim3(N / TILE_N, M / TILE_M), 256, 0, stream>>>(
        pn, mn, nullptr, out, M, N, D, temperature);
}
